// Codebook_85933705658932
// MI455X (gfx1250) — matrix operand order repaired and checked by reading, not compiled, not run on hardware
//
#include <hip/hip_runtime.h>
#include <hip/hip_bf16.h>
#include <stdint.h>

// ---------------------------------------------------------------------------
// VQ codebook quantization for MI455X (gfx1250, wave32, WMMA + TDM).
// cross = xt @ codes^T via v_wmma_f32_16x16x32_bf16 (compute-bound: 34.4 GFLOP
// vs ~70MB HBM). Codebook chunks are staged into LDS with the Tensor Data
// Mover (tensor_load_to_lds, TENSORcnt) and shared by all 8 waves, cutting L2
// read traffic ~8x vs per-wave global streaming. The TDM descriptor's LDS
// padding (interval=128 DW, amount=4 DW -> 528B row stride) makes the
// per-lane ds_load_b128 B-fragment reads bank-conflict free.
// argmin uses score = |c|^2 - 2*cross (the |x|^2 term is per-pixel constant).
// ---------------------------------------------------------------------------

typedef __attribute__((ext_vector_type(16))) __bf16 v16bf;
typedef __attribute__((ext_vector_type(8)))  float  v8f;
typedef __attribute__((ext_vector_type(4)))  unsigned int u32x4;
typedef __attribute__((ext_vector_type(8)))  int i32x8;
typedef __attribute__((ext_vector_type(4)))  int i32x4;

#define B_    32
#define D_    256
#define HW_   1024
#define K_    2048
#define NPIX  (B_ * HW_)         // 32768
#define XE_ELEMS (B_ * D_ * HW_) // 8388608

#define CH        32             // codes per LDS chunk
#define NCH       (K_ / CH)      // 64 chunks
#define ROW_SH    264            // LDS row stride in shorts (512B data + 16B pad)

#if defined(__has_builtin)
#if __has_builtin(__builtin_amdgcn_tensor_load_to_lds) && \
    __has_builtin(__builtin_amdgcn_s_wait_tensorcnt)
#define USE_TDM 1
#endif
#endif

static __device__ __forceinline__ unsigned short f2bf(float f) {
    unsigned u = __float_as_uint(f);
    u = u + 0x7fffu + ((u >> 16) & 1u);   // round-to-nearest-even
    return (unsigned short)(u >> 16);
}

// --- Kernel 1: codes -> bf16 (row-major [k][d]) + |c|^2 ---------------------
__global__ __launch_bounds__(256)
void prep_codes_kernel(const float* __restrict__ lut,
                       unsigned short* __restrict__ cb,
                       float* __restrict__ csq) {
    __shared__ float red[256];
    const int k = blockIdx.x, t = threadIdx.x;
    float v = lut[(size_t)k * D_ + t];
    cb[(size_t)k * D_ + t] = f2bf(v);
    red[t] = v * v;
    __syncthreads();
    for (int s = 128; s > 0; s >>= 1) {
        if (t < s) red[t] += red[t + s];
        __syncthreads();
    }
    if (t == 0) csq[k] = red[0];
}

// --- Kernel 2: x (b,d,hw) -> xt bf16 (b*hw, d), LDS 32x32 tile transpose ----
__global__ __launch_bounds__(256)
void prep_x_kernel(const float* __restrict__ x, unsigned short* __restrict__ xt) {
    __shared__ float tile[32][33];
    const int b = blockIdx.z, d0 = blockIdx.y * 32, n0 = blockIdx.x * 32;
    const int tx = threadIdx.x & 31, ty = threadIdx.x >> 5;  // 32 x 8
    const float* xb = x + (size_t)b * (D_ * HW_);
#pragma unroll
    for (int i = 0; i < 4; ++i) {
        int dl = ty + i * 8;
        tile[dl][tx] = xb[(size_t)(d0 + dl) * HW_ + n0 + tx];  // coalesced
    }
    __syncthreads();
    unsigned short* xo = xt + (size_t)b * HW_ * D_;
#pragma unroll
    for (int i = 0; i < 4; ++i) {
        int nl = ty + i * 8;
        xo[(size_t)(n0 + nl) * D_ + d0 + tx] = f2bf(tile[tx][nl]);  // coalesced
    }
}

// --- TDM issue: one 2D tile (CH rows x 512B) -> padded LDS rows -------------
#if defined(USE_TDM)
static __device__ __forceinline__ void tdm_issue(const unsigned short* gsrc,
                                                 unsigned lds_off) {
    const unsigned long long ga = (unsigned long long)(size_t)gsrc;
    u32x4 g0;
    g0[0] = 1u;                                  // count=1, user descriptor
    g0[1] = lds_off;                             // lds_addr (bytes)
    g0[2] = (unsigned)(ga & 0xffffffffu);        // global_addr[31:0]
    g0[3] = (unsigned)((ga >> 32) & 0x01ffffffu) // global_addr[56:32]
          | (2u << 30);                          // type=2 ("image")
    i32x8 g1;
    // workgroup_mask=0 | data_size=4B | pad_enable | pad_interval=128DW(code 6)
    // | pad_amount=4DW(code 3)
    g1[0] = (int)0x07920000u;
    g1[1] = (int)(128u << 16);   // tensor_dim0 = 128 (4B units) [bits 79:48]
    g1[2] = (int)((unsigned)CH << 16);  // tensor_dim1 = CH      [bits 111:80]
    g1[3] = (int)(128u << 16);   // tile_dim0 = 128              [bits 127:112]
    g1[4] = CH;                  // tile_dim1 = CH, tile_dim2=0   [bits 159:128]
    g1[5] = 128;                 // tensor_dim0_stride = 128      [bits 207:160]
    g1[6] = 0;
    g1[7] = 0;
    const i32x4 gz4 = {0, 0, 0, 0};
#if __clang_major__ >= 23
    const i32x8 gz8 = {0, 0, 0, 0, 0, 0, 0, 0};
    __builtin_amdgcn_tensor_load_to_lds(g0, g1, gz4, gz4, gz8, 0);
#else
    __builtin_amdgcn_tensor_load_to_lds(g0, g1, gz4, gz4, 0);
#endif
}
#endif

// --- Kernel 3: WMMA GEMM + fused argmin ------------------------------------
// Block = 256 thr = 8 waves; tile = 128 pixels x all 2048 codes.
// Wave w owns pixels [blk*128 + w*16, +16); A fragments (64 VGPRs) resident.
// Codebook streamed through double-buffered LDS chunks loaded by the TDM.
union FragU { v16bf v; uint4 q[2]; };

__global__ __launch_bounds__(256)
void gemm_argmin_kernel(const unsigned short* __restrict__ xt,
                        const unsigned short* __restrict__ cb,
                        const float* __restrict__ csq,
                        int* __restrict__ qout,
                        float* __restrict__ qf) {
    __shared__ unsigned short lbuf[2][CH * ROW_SH];   // 2 x 16,896 B

    const int tid  = threadIdx.x;
    const int wave = tid >> 5;
    const int lane = tid & 31;
    const int col  = lane & 15;
    const int half = lane >> 4;
    const int row0 = blockIdx.x * 128 + wave * 16;

#if defined(USE_TDM)
    if (wave == 0) tdm_issue(cb, (unsigned)(size_t)&lbuf[0][0]);
#endif

    // A fragment (ISA layout): lane holds row (lane&15); lanes<16 K in
    // {0..7,16..23}, lanes>=16 K in {8..15,24..31} of each 32-wide d-chunk.
    FragU afrag[8];
    const unsigned short* arow = xt + (size_t)(row0 + col) * D_;
#pragma unroll
    for (int kf = 0; kf < 8; ++kf) {
        const int dbase = kf * 32;
        afrag[kf].q[0] = *(const uint4*)(arow + dbase + half * 8);
        afrag[kf].q[1] = *(const uint4*)(arow + dbase + 16 + half * 8);
    }

    float best[8];
    int   bestk[8];
#pragma unroll
    for (int r = 0; r < 8; ++r) { best[r] = 3.4e38f; bestk[r] = 0; }

#if defined(USE_TDM)
    if (wave == 0) __builtin_amdgcn_s_wait_tensorcnt(0);
    __syncthreads();
#endif

    for (int ch = 0; ch < NCH; ++ch) {
#if defined(USE_TDM)
        const int buf = ch & 1;
        if (wave == 0 && (ch + 1) < NCH)   // prefetch next chunk (other buffer)
            tdm_issue(cb + (size_t)(ch + 1) * CH * D_,
                      (unsigned)(size_t)&lbuf[buf ^ 1][0]);
#else
        const int buf = 0;
        __syncthreads();                   // previous chunk's compute done
        {   // cooperative copy: CH rows x 32 uint4; 256 thr x 4 uint4 each
            const int row = tid >> 3, seg = (tid & 7) * 4;
            const uint4* src = (const uint4*)(cb + ((size_t)ch * CH + row) * D_);
#pragma unroll
            for (int i = 0; i < 4; ++i)
                *(uint4*)&lbuf[0][row * ROW_SH + (seg + i) * 8] = src[seg + i];
        }
        __syncthreads();
#endif
        const unsigned short* bbase = &lbuf[buf][0];
#pragma unroll
        for (int nt = 0; nt < CH / 16; ++nt) {
            const int klocal = nt * 16 + col;
            const int k = ch * CH + klocal;
            const unsigned short* brow = bbase + klocal * ROW_SH;
            v8f acc = {0.f, 0.f, 0.f, 0.f, 0.f, 0.f, 0.f, 0.f};
#pragma unroll
            for (int kf = 0; kf < 8; ++kf) {
                // B fragment: lane = column; lanes<16 K rows {0..7,16..23},
                // lanes>=16 K rows {8..15,24..31} of each 32-wide d-chunk.
                FragU bfrag;
                bfrag.q[0] = *(const uint4*)(brow + kf * 32 + half * 8);
                bfrag.q[1] = *(const uint4*)(brow + kf * 32 + 16 + half * 8);
                acc = __builtin_amdgcn_wmma_f32_16x16x32_bf16(
                    false, afrag[kf].v, false, bfrag.v, (short)0, acc, false, false);
            }
            const float cs = csq[k];
#pragma unroll
            for (int r = 0; r < 8; ++r) {          // C: lanes<16 M=r, >=16 M=r+8
                float s = cs - 2.0f * acc[r];
                if (s < best[r]) { best[r] = s; bestk[r] = k; }
            }
        }
#if defined(USE_TDM)
        if (wave == 0) __builtin_amdgcn_s_wait_tensorcnt(0);
        __syncthreads();                   // next chunk visible to all waves
#endif
    }

    // Cross-lane argmin butterfly within each 16-lane half (rows 0..7 in
    // lanes 0..15, rows 8..15 in lanes 16..31); first-index tie-break.
#pragma unroll
    for (int m = 1; m <= 8; m <<= 1) {
#pragma unroll
        for (int r = 0; r < 8; ++r) {
            float os = __shfl_xor(best[r], m, 32);
            int   ok = __shfl_xor(bestk[r], m, 32);
            if (os < best[r] || (os == best[r] && ok < bestk[r])) {
                best[r] = os; bestk[r] = ok;
            }
        }
    }
    if (col == 0) {                        // lane 0 -> rows 0..7, lane 16 -> 8..15
        const int base = row0 + half * 8;
#pragma unroll
        for (int r = 0; r < 8; ++r) {
            qout[base + r] = bestk[r];
            qf[base + r]   = (float)bestk[r];
        }
    }
}

// --- Kernel 4: gather x_e (coalesced store) + loss partials -----------------
// x_for_loss[b, flat i] = x[b][i%256][i/256]; x_e[b, flat i] = lut[q[b][i%1024]][i/1024].
__global__ __launch_bounds__(256)
void gather_loss_kernel(const float* __restrict__ x, const float* __restrict__ lut,
                        const int* __restrict__ q, float* __restrict__ xe,
                        float* __restrict__ partials) {
    __shared__ float red[256];
    const int tid = threadIdx.x;
    float acc = 0.f;
    const size_t base   = (size_t)blockIdx.x * 256 + tid;
    const size_t stride = (size_t)gridDim.x * 256;
#pragma unroll
    for (int it = 0; it < 8; ++it) {
        const size_t idx = base + (size_t)it * stride;     // 4096*256*8 == XE_ELEMS
        const int b  = (int)(idx >> 18);                   // / 262144
        const int j  = (int)(idx & 262143);
        const int dd = j >> 10;
        const int n  = j & 1023;
        const int qq = q[b * HW_ + n];
        const float vb = lut[(size_t)qq * D_ + dd];
        const float va = x[(size_t)b * (D_ * HW_) + (size_t)(j & 255) * HW_ + (j >> 8)];
        xe[idx] = vb;
        const float df = va - vb;
        acc += df * df;
    }
    red[tid] = acc;
    __syncthreads();
    for (int s = 128; s > 0; s >>= 1) {
        if (tid < s) red[tid] += red[tid + s];
        __syncthreads();
    }
    if (tid == 0) partials[blockIdx.x] = red[0];
}

// --- Kernel 5: deterministic final reduction -------------------------------
__global__ __launch_bounds__(256)
void final_loss_kernel(const float* __restrict__ partials, float* __restrict__ lossout) {
    __shared__ float red[256];
    const int t = threadIdx.x;
    float s = 0.f;
#pragma unroll
    for (int i = 0; i < 16; ++i) s += partials[t + 256 * i];
    red[t] = s;
    __syncthreads();
    for (int st = 128; st > 0; st >>= 1) {
        if (t < st) red[t] += red[t + st];
        __syncthreads();
    }
    if (t == 0) lossout[0] = red[0] / (float)XE_ELEMS;
}

// ---------------------------------------------------------------------------
extern "C" void kernel_launch(void* const* d_in, const int* in_sizes, int n_in,
                              void* d_out, int out_size, void* d_ws, size_t ws_size,
                              hipStream_t stream) {
    const float* x   = (const float*)d_in[0];   // (32,256,32,32)
    const float* lut = (const float*)d_in[1];   // (1,1,2048,256)
    float* out = (float*)d_out;                 // [x_e | q_x | loss]
    char*  ws  = (char*)d_ws;

    unsigned short* cb  = (unsigned short*)(ws);              // 2048*256*2 = 1 MB
    float*          csq = (float*)(ws + 1048576);             // 8 KB
    unsigned short* xt  = (unsigned short*)(ws + 1056768);    // 32768*256*2 = 16 MB
    int*            q   = (int*)(ws + 17833984);              // 128 KB
    float*          par = (float*)(ws + 17965056);            // 16 KB

    prep_codes_kernel<<<K_, 256, 0, stream>>>(lut, cb, csq);
    prep_x_kernel<<<dim3(HW_ / 32, D_ / 32, B_), 256, 0, stream>>>(x, xt);
    gemm_argmin_kernel<<<NPIX / 128, 256, 0, stream>>>(xt, cb, csq, q, out + XE_ELEMS);
    gather_loss_kernel<<<4096, 256, 0, stream>>>(x, lut, q, out, par);
    final_loss_kernel<<<1, 256, 0, stream>>>(par, out + XE_ELEMS + NPIX);
}
